// MultiHeadAttention_23914377904814
// MI455X (gfx1250) — compile-verified, matrix operand order checked by reading, not run on hardware
//
#include <hip/hip_runtime.h>

typedef _Float16 half8  __attribute__((ext_vector_type(8)));
typedef _Float16 half16 __attribute__((ext_vector_type(16)));
typedef float    float8 __attribute__((ext_vector_type(8)));

constexpr int Bn  = 4;
constexpr int Sn  = 2048;
constexpr int Dn  = 1024;
constexpr int Hn  = 16;
constexpr int DHn = 64;

__device__ inline half16 hcat(half8 lo, half8 hi) {
    return __builtin_shufflevector(lo, hi, 0,1,2,3,4,5,6,7,8,9,10,11,12,13,14,15);
}

__device__ inline float8 wmma_f16(half16 a, half16 b, float8 c) {
    return __builtin_amdgcn_wmma_f32_16x16x32_f16(
        /*neg_a=*/false, a, /*neg_b=*/false, b,
        /*c_mod=*/(short)0, c, /*reuse_a=*/false, /*reuse_b=*/false);
}

// Per-lane 16-byte async global->LDS copy (ASYNCcnt-tracked).
__device__ inline void async_copy_b128(uint32_t lds_byte_addr, uint64_t global_addr) {
    asm volatile("global_load_async_to_lds_b128 %0, %1, off"
                 :: "v"(lds_byte_addr), "v"(global_addr) : "memory");
}
__device__ inline void wait_asynccnt0() {
    asm volatile("s_wait_asynccnt 0x0" ::: "memory");
}

__device__ inline half8 cvt8(const float* s) {
    half8 h;
#pragma unroll
    for (int i = 0; i < 8; ++i) h[i] = (_Float16)s[i];
    return h;
}

// ---------------------------------------------------------------------------
// Kernel 1: Y = X @ W + bias  (X: [B*S, D] f32, W: [D, D] f32, bias: [D])
// Output f16 in head-major layout [B, H, S, DH].
// Block tile M=64 x N=128, K-step 32. 8 waves: wave w owns N-subtile w and
// computes 4 M-subtiles, reusing its B-fragment across 4 WMMAs per k-step.
// ---------------------------------------------------------------------------
__global__ __launch_bounds__(256) void qkv_proj_kernel(
    const float* __restrict__ X, const float* __restrict__ W,
    const float* __restrict__ bias, _Float16* __restrict__ Out)
{
    __shared__ __align__(16) _Float16 As[64 * 32];   // 4 KB
    __shared__ __align__(16) _Float16 Bs[32 * 128];  // 8 KB

    const int t       = threadIdx.x;
    const int wave    = t >> 5;
    const int lane    = t & 31;
    const int half_id = lane >> 4;
    const int lr      = lane & 15;

    const int m0  = blockIdx.x * 64;
    const int nb  = blockIdx.y * 128;
    const int col = wave * 16 + lr;          // column within block tile

    float8 acc[4] = {};
    for (int k0 = 0; k0 < Dn; k0 += 32) {
        __syncthreads();
        {   // stage A tile 64x32: 8 consecutive f32 per thread, coalesced
            const int r  = t >> 2;           // 0..63
            const int kk = (t & 3) * 8;      // 0,8,16,24
            const float* src = X + (size_t)(m0 + r) * Dn + k0 + kk;
            *(half8*)(As + r * 32 + kk) = cvt8(src);
        }
        {   // stage B tile 32x128: 16 consecutive f32 per thread, coalesced
            const int r  = t >> 3;           // 0..31
            const int c0 = (t & 7) * 16;     // 0..112
            const float* src = W + (size_t)(k0 + r) * Dn + nb + c0;
            *(half8*)(Bs + r * 128 + c0)     = cvt8(src);
            *(half8*)(Bs + r * 128 + c0 + 8) = cvt8(src + 8);
        }
        __syncthreads();

        half16 b;
#pragma unroll
        for (int i = 0; i < 8; ++i) {
            b[i]     = Bs[(8 * half_id + i) * 128 + col];
            b[8 + i] = Bs[(16 + 8 * half_id + i) * 128 + col];
        }
#pragma unroll
        for (int mt = 0; mt < 4; ++mt) {
            const _Float16* arow = As + (mt * 16 + lr) * 32;
            const half16 a = hcat(*(const half8*)(arow + 8 * half_id),
                                  *(const half8*)(arow + 16 + 8 * half_id));
            acc[mt] = wmma_f16(a, b, acc[mt]);
        }
    }

    const int   nn  = nb + col;
    const float bvs = bias[nn];
    const int   h   = nn >> 6;
    const int   dh  = nn & 63;
#pragma unroll
    for (int mt = 0; mt < 4; ++mt) {
#pragma unroll
        for (int v = 0; v < 8; ++v) {
            const int m   = m0 + mt * 16 + v + 8 * half_id;
            const int bbi = m >> 11;         // m / Sn
            const int s   = m & (Sn - 1);    // m % Sn
            Out[(((size_t)bbi * Hn + h) * Sn + s) * DHn + dh] =
                (_Float16)(acc[mt][v] + bvs);
        }
    }
}

// ---------------------------------------------------------------------------
// Kernel 2: flash attention. One (b,h) slice per block; 8 waves x 16 q-rows.
// 64-key blocks double-buffered in LDS via async global->LDS DMA; per
// iteration each wave runs 16 WMMAs (8 QK^T + 8 PV) against one softmax pass.
// ---------------------------------------------------------------------------
__global__ __launch_bounds__(256) void flash_attn_kernel(
    const _Float16* __restrict__ Qh, const _Float16* __restrict__ Kh,
    const _Float16* __restrict__ Vh, const unsigned char* __restrict__ mask,
    _Float16* __restrict__ Ah)
{
    constexpr int KB = 64;                       // keys per block-iteration
    __shared__ __align__(16) _Float16 Klds[2][KB * DHn];   // 2 x 8 KB
    __shared__ __align__(16) _Float16 Vlds[2][KB * DHn];   // 2 x 8 KB
    __shared__ __align__(16) _Float16 Plds[8][16 * KB];    // 16 KB

    const int t       = threadIdx.x;
    const int wave    = t >> 5;
    const int lane    = t & 31;
    const int half_id = lane >> 4;
    const int lr      = lane & 15;

    const int bb = blockIdx.z;
    const int h  = blockIdx.y;
    const int q0 = (blockIdx.x * 8 + wave) * 16;

    const size_t headoff = ((size_t)bb * Hn + h) * Sn * DHn;

    // Q fragments (row q0+lr): K-dim 64 = two 16x32 A-frags, kept in VGPRs
    const _Float16* qrow = Qh + headoff + (size_t)(q0 + lr) * DHn;
    const half16 qa0 = hcat(*(const half8*)(qrow +      8 * half_id),
                            *(const half8*)(qrow + 16 + 8 * half_id));
    const half16 qa1 = hcat(*(const half8*)(qrow + 32 + 8 * half_id),
                            *(const half8*)(qrow + 48 + 8 * half_id));

    float8 oAcc[4] = {};
    float mrow[8], lrow[8];
#pragma unroll
    for (int v = 0; v < 8; ++v) { mrow[v] = -1e30f; lrow[v] = 0.0f; }

    const unsigned char* mptr = mask + (size_t)bb * Sn;
    const float NEGBIG = -1e30f;

    // async DMA of 64x64 f16 K and V tiles into buffer (jblk & 1): 32B/lane each
    auto issue_block = [&](int jblk) {
        const int buf = jblk & 1;
        const uint64_t ksrc = (uint64_t)(uintptr_t)(Kh + headoff + (size_t)jblk * KB * DHn) + t * 16u;
        const uint64_t vsrc = (uint64_t)(uintptr_t)(Vh + headoff + (size_t)jblk * KB * DHn) + t * 16u;
        const uint32_t kdst = (uint32_t)(uintptr_t)(&Klds[buf][0]) + t * 16u;
        const uint32_t vdst = (uint32_t)(uintptr_t)(&Vlds[buf][0]) + t * 16u;
        async_copy_b128(kdst, ksrc);
        async_copy_b128(kdst + 4096u, ksrc + 4096u);
        async_copy_b128(vdst, vsrc);
        async_copy_b128(vdst + 4096u, vsrc + 4096u);
    };

    constexpr int NB = Sn / KB;                 // 32 key blocks
    issue_block(0);

    for (int jblk = 0; jblk < NB; ++jblk) {
        const int j0  = jblk * KB;
        const int cur = jblk & 1;

        wait_asynccnt0();                       // this wave's DMA for `cur` done
        __syncthreads();                        // all chunks visible; prev compute done
        if (jblk + 1 < NB) issue_block(jblk + 1);   // overlap next DMA with compute

        const _Float16* Kb = Klds[cur];
        const _Float16* Vb = Vlds[cur];

        // scores: 4 x 16-key C-tiles, each 2 chained WMMAs over K-dim 64
        float8 sc[4];
#pragma unroll
        for (int g = 0; g < 4; ++g) {
            const _Float16* kr = Kb + (size_t)(g * 16 + lr) * DHn;
            const half16 bk0 = hcat(*(const half8*)(kr +      8 * half_id),
                                    *(const half8*)(kr + 16 + 8 * half_id));
            const half16 bk1 = hcat(*(const half8*)(kr + 32 + 8 * half_id),
                                    *(const half8*)(kr + 48 + 8 * half_id));
            float8 c = {};
            c = wmma_f16(qa0, bk0, c);
            c = wmma_f16(qa1, bk1, c);
            sc[g] = c;
        }

        // scale + key-padding mask (column = key index per lane)
#pragma unroll
        for (int g = 0; g < 4; ++g) {
            const bool mk = mptr[j0 + g * 16 + lr] != 0;
#pragma unroll
            for (int v = 0; v < 8; ++v)
                sc[g][v] = mk ? NEGBIG : sc[g][v] * 0.125f;   // 1/sqrt(DH)
        }

        // online softmax: one reduction pass covers all 64 keys
#pragma unroll
        for (int v = 0; v < 8; ++v) {
            float x = fmaxf(fmaxf(sc[0][v], sc[1][v]), fmaxf(sc[2][v], sc[3][v]));
            x = fmaxf(x, __shfl_xor(x, 1));
            x = fmaxf(x, __shfl_xor(x, 2));
            x = fmaxf(x, __shfl_xor(x, 4));
            x = fmaxf(x, __shfl_xor(x, 8));
            const float mnew = fmaxf(mrow[v], x);
            const float corr = __expf(mrow[v] - mnew);
            mrow[v] = mnew;
            lrow[v] *= corr;
            oAcc[0][v] *= corr; oAcc[1][v] *= corr;
            oAcc[2][v] *= corr; oAcc[3][v] *= corr;
            float rs = 0.0f;
#pragma unroll
            for (int g = 0; g < 4; ++g) {
                const float p = __expf(sc[g][v] - mnew);
                sc[g][v] = p;
                rs += p;
            }
            rs += __shfl_xor(rs, 1);
            rs += __shfl_xor(rs, 2);
            rs += __shfl_xor(rs, 4);
            rs += __shfl_xor(rs, 8);
            lrow[v] += rs;
        }

        // C-layout -> A-layout swizzle of P through per-wave LDS tile (16x64 f16)
        _Float16* P = Plds[wave];
#pragma unroll
        for (int g = 0; g < 4; ++g)
#pragma unroll
            for (int v = 0; v < 8; ++v)
                P[(v + 8 * half_id) * KB + g * 16 + lr] = (_Float16)sc[g][v];
        asm volatile("s_wait_dscnt 0x0" ::: "memory");   // same-wave DS RAW fence
        const _Float16* prow = P + (size_t)lr * KB;
        const half16 pa0 = hcat(*(const half8*)(prow +      8 * half_id),
                                *(const half8*)(prow + 16 + 8 * half_id));
        const half16 pa1 = hcat(*(const half8*)(prow + 32 + 8 * half_id),
                                *(const half8*)(prow + 48 + 8 * half_id));

        // PV: per dh-tile, B-frags are V columns over keys 0..31 / 32..63
#pragma unroll
        for (int dt = 0; dt < 4; ++dt) {
            half16 bvlo, bvhi;
#pragma unroll
            for (int i = 0; i < 8; ++i) {
                const int kA = (8 * half_id + i) * DHn + dt * 16 + lr;
                const int kBi = (16 + 8 * half_id + i) * DHn + dt * 16 + lr;
                bvlo[i] = Vb[kA];            bvlo[8 + i] = Vb[kBi];
                bvhi[i] = Vb[kA + 32 * DHn]; bvhi[8 + i] = Vb[kBi + 32 * DHn];
            }
            oAcc[dt] = wmma_f16(pa0, bvlo, oAcc[dt]);
            oAcc[dt] = wmma_f16(pa1, bvhi, oAcc[dt]);
        }
    }

    // normalize and write attn output as f16 in [B, S, H*DH] layout
#pragma unroll
    for (int v = 0; v < 8; ++v) {
        const float inv = 1.0f / lrow[v];
        const int   s   = q0 + v + 8 * half_id;
        _Float16* orow = Ah + ((size_t)(bb * Sn + s) * Dn) + h * DHn;
#pragma unroll
        for (int dt = 0; dt < 4; ++dt)
            orow[dt * 16 + lr] = (_Float16)(oAcc[dt][v] * inv);
    }
}

// ---------------------------------------------------------------------------
// Kernel 3: out = Ah @ Wo + bo   (Ah: [B*S, D] f16, Wo: [D, D] f32, out f32)
// Same M=64 x N=128 staged tiling as kernel 1 (A already f16).
// ---------------------------------------------------------------------------
__global__ __launch_bounds__(256) void out_proj_kernel(
    const _Float16* __restrict__ A, const float* __restrict__ W,
    const float* __restrict__ bias, float* __restrict__ Out)
{
    __shared__ __align__(16) _Float16 As[64 * 32];
    __shared__ __align__(16) _Float16 Bs[32 * 128];

    const int t       = threadIdx.x;
    const int wave    = t >> 5;
    const int lane    = t & 31;
    const int half_id = lane >> 4;
    const int lr      = lane & 15;

    const int m0  = blockIdx.x * 64;
    const int nb  = blockIdx.y * 128;
    const int col = wave * 16 + lr;

    float8 acc[4] = {};
    for (int k0 = 0; k0 < Dn; k0 += 32) {
        __syncthreads();
        {   // stage A tile 64x32 f16: one b128 per thread
            const int r  = t >> 2;            // 0..63
            const int cc = (t & 3) * 8;       // 0,8,16,24
            *(half8*)(As + r * 32 + cc) =
                *(const half8*)(A + (size_t)(m0 + r) * Dn + k0 + cc);
        }
        {   // stage B tile 32x128 f32 -> f16, coalesced
            const int r  = t >> 3;
            const int c0 = (t & 7) * 16;
            const float* src = W + (size_t)(k0 + r) * Dn + nb + c0;
            *(half8*)(Bs + r * 128 + c0)     = cvt8(src);
            *(half8*)(Bs + r * 128 + c0 + 8) = cvt8(src + 8);
        }
        __syncthreads();

        half16 b;
#pragma unroll
        for (int i = 0; i < 8; ++i) {
            b[i]     = Bs[(8 * half_id + i) * 128 + col];
            b[8 + i] = Bs[(16 + 8 * half_id + i) * 128 + col];
        }
#pragma unroll
        for (int mt = 0; mt < 4; ++mt) {
            const _Float16* arow = As + (mt * 16 + lr) * 32;
            const half16 a = hcat(*(const half8*)(arow + 8 * half_id),
                                  *(const half8*)(arow + 16 + 8 * half_id));
            acc[mt] = wmma_f16(a, b, acc[mt]);
        }
    }

    const int   nn  = nb + col;
    const float bvs = bias[nn];
#pragma unroll
    for (int mt = 0; mt < 4; ++mt)
#pragma unroll
        for (int v = 0; v < 8; ++v) {
            const int m = m0 + mt * 16 + v + 8 * half_id;
            Out[(size_t)m * Dn + nn] = acc[mt][v] + bvs;
        }
}

// ---------------------------------------------------------------------------
extern "C" void kernel_launch(void* const* d_in, const int* in_sizes, int n_in,
                              void* d_out, int out_size, void* d_ws, size_t ws_size,
                              hipStream_t stream) {
    const float* query = (const float*)d_in[0];
    const float* key   = (const float*)d_in[1];
    const float* value = (const float*)d_in[2];
    const unsigned char* mask = (const unsigned char*)d_in[3];  // numpy bool (1B)
    const float* Wq = (const float*)d_in[4];
    const float* bq = (const float*)d_in[5];
    const float* Wk = (const float*)d_in[6];
    const float* bk = (const float*)d_in[7];
    const float* Wv = (const float*)d_in[8];
    const float* bv = (const float*)d_in[9];
    const float* Wo = (const float*)d_in[10];
    const float* bo = (const float*)d_in[11];

    const size_t headElems = (size_t)Bn * Hn * Sn * DHn;  // 8.39M f16 each
    _Float16* Qh = (_Float16*)d_ws;
    _Float16* Kh = Qh + headElems;
    _Float16* Vh = Kh + headElems;
    _Float16* Ah = Vh + headElems;

    const dim3 gproj(Bn * Sn / 64, Dn / 128);   // (128, 8): M=64 x N=128 tiles
    qkv_proj_kernel<<<gproj, 256, 0, stream>>>(query, Wq, bq, Qh);
    qkv_proj_kernel<<<gproj, 256, 0, stream>>>(key,   Wk, bk, Kh);
    qkv_proj_kernel<<<gproj, 256, 0, stream>>>(value, Wv, bv, Vh);

    flash_attn_kernel<<<dim3(Sn / 128, Hn, Bn), 256, 0, stream>>>(Qh, Kh, Vh, mask, Ah);

    out_proj_kernel<<<gproj, 256, 0, stream>>>(Ah, Wo, bo, (float*)d_out);
}
